// GATWithSentenceEmbedding_3221225472737
// MI455X (gfx1250) — matrix operand order repaired and checked by reading, not compiled, not run on hardware
//
#include <hip/hip_runtime.h>
#include <hip/hip_bf16.h>
#include <hip/hip_fp16.h>

// ---------------------------------------------------------------------------
// Types for CDNA5 WMMA (wave32, 16x16x32 f16 -> f32)
// ---------------------------------------------------------------------------
typedef _Float16 f16_t;
typedef __attribute__((ext_vector_type(16))) _Float16 v16h_t;
typedef __attribute__((ext_vector_type(8)))  float    v8f_t;
typedef __attribute__((ext_vector_type(2)))  int      v2i_t;

union Frag16 { v16h_t v; unsigned u[8]; };

// Async global->LDS staging (CDNA5): use when the toolchain declares the
// builtins; otherwise fall back to register ping-pong staging.
// Probe-confirmed: builtin exists here with signature (v2i AS1*, v2i AS3*, Ii, Ii).
#if defined(__HIP_DEVICE_COMPILE__) && defined(__has_builtin)
#if __has_builtin(__builtin_amdgcn_global_load_async_to_lds_b64) && \
    __has_builtin(__builtin_amdgcn_s_wait_asynccnt)
#define USE_ASYNC_LDS 1
#endif
#endif
#ifndef USE_ASYNC_LDS
#define USE_ASYNC_LDS 0
#endif

#if USE_ASYNC_LDS
#define AS_GLOBAL __attribute__((address_space(1)))
#define AS_LOCAL  __attribute__((address_space(3)))
#endif

// Order-preserving float<->uint mapping for atomicMax-based segment max
__device__ __forceinline__ unsigned f2ord(float f) {
  unsigned u = __float_as_uint(f);
  return (u & 0x80000000u) ? ~u : (u | 0x80000000u);
}
__device__ __forceinline__ float ord2f(unsigned o) {
  unsigned u = (o & 0x80000000u) ? (o & 0x7FFFFFFFu) : ~o;
  return __uint_as_float(u);
}

__device__ __forceinline__ void atomic_fadd(float* p, float v) {
  __hip_atomic_fetch_add(p, v, __ATOMIC_RELAXED, __HIP_MEMORY_SCOPE_AGENT);
}

// ---------------------------------------------------------------------------
// f32 -> f16 conversion / transpose helpers
// ---------------------------------------------------------------------------
__global__ void f32_to_f16_kernel(const float* __restrict__ in,
                                  f16_t* __restrict__ out, size_t n) {
  size_t t = (size_t)blockIdx.x * blockDim.x + threadIdx.x;
  if (t < n) out[t] = (f16_t)in[t];
}

// Wt[j*K + k] = (f16) W[k*Ncols + j]   (W row-major [K, Ncols])
__global__ void transpose_f16_kernel(const float* __restrict__ W,
                                     f16_t* __restrict__ Wt, int K, int Ncols) {
  int t = blockIdx.x * blockDim.x + threadIdx.x;
  if (t >= K * Ncols) return;
  int j = t / K;
  int k = t - j * K;
  Wt[t] = (f16_t)W[(size_t)k * Ncols + j];
}

// sp[j] = sum_k sent[k] * W1[(768+k)*512 + j]   (sentence half of GEMM1, once)
__global__ void sent_proj_kernel(const float* __restrict__ sent,
                                 const float* __restrict__ W1,
                                 float* __restrict__ sp) {
  int j = blockIdx.x * blockDim.x + threadIdx.x;
  if (j >= 512) return;
  float acc = 0.f;
  for (int k = 0; k < 768; ++k)
    acc = fmaf(sent[k], W1[(size_t)(768 + k) * 512 + j], acc);
  sp[j] = acc;
}

__global__ void fill_bias_kernel(float* __restrict__ out,
                                 const float* __restrict__ bias,
                                 int rows, int cols) {
  int t = blockIdx.x * blockDim.x + threadIdx.x;
  if (t < rows * cols) out[t] = bias[t % cols];
}

__global__ void fill_ms_kernel(unsigned* __restrict__ m, float* __restrict__ s, int n) {
  int t = blockIdx.x * blockDim.x + threadIdx.x;
  if (t < n) { m[t] = 0u; s[t] = 0.f; }  // ord(0) < ord(any finite/-inf float)
}

// ---------------------------------------------------------------------------
// Tiled WMMA GEMM:  C[M,N] = A[M,K](f16) * Bt[N,K]^T(f16) (+ colBias[N])
//   Block tile 32(M) x WN*64(N), K-step 32, double-buffered LDS with async
//   global->LDS DMA (ASYNCcnt) when available.
//   WN waves; each wave owns a 32x64 strip = 8 WMMA accumulators, so two
//   A fragments are amortized over four B fragments per K-tile.
//   Requires M%32==0, N%(WN*64)==0, K%32==0 (20000 / 512 / 768,512: all ok).
// ---------------------------------------------------------------------------
template <int WN>
__global__ __launch_bounds__(WN * 32) void gemm_f16_wmma_kernel(
    const f16_t* __restrict__ A, const f16_t* __restrict__ Bt,
    float* __restrict__ C, const float* __restrict__ colBias,
    int M, int N, int K) {
  constexpr int BM = 32;
  constexpr int BN = WN * 64;
  constexpr int BK = 32;
  constexpr int LS = 36;       // u16 per LDS row (pad: breaks bank conflicts)
  constexpr int T  = WN * 32;  // threads per block
  constexpr int NCA = 256 / T; // b64 chunks per thread for the A tile

  __shared__ __align__(16) unsigned short As[2][BM * LS];
  __shared__ __align__(16) unsigned short Bs[2][BN * LS];

  const int tid   = threadIdx.x;
  const int lane  = tid & 31;
  const int waveN = tid >> 5;  // 0..WN-1
  const int blockRow = blockIdx.x * BM;
  const int blockCol = blockIdx.y * BN;

  const unsigned short* Au = (const unsigned short*)A + (size_t)blockRow * K;
  const unsigned short* Bu = (const unsigned short*)Bt + (size_t)blockCol * K;

  v8f_t acc[2][4];
#pragma unroll
  for (int i = 0; i < 2; ++i)
#pragma unroll
    for (int j = 0; j < 4; ++j)
#pragma unroll
      for (int v = 0; v < 8; ++v) acc[i][j][v] = 0.f;

#if !USE_ASYNC_LDS
  unsigned long long regA[NCA];
  unsigned long long regB[8];
#endif

  // Issue the global-side loads for K-tile t into LDS buffer `buf`
  auto issue = [&](int t, int buf) {
    const int k0 = t * BK;
#pragma unroll
    for (int i = 0; i < NCA; ++i) {
      int idx = (i * T + tid) * 4;
      int r = idx >> 5, c = idx & 31;
#if USE_ASYNC_LDS
      __builtin_amdgcn_global_load_async_to_lds_b64(
          (AS_GLOBAL v2i_t*)(Au + (size_t)r * K + k0 + c),
          (AS_LOCAL v2i_t*)&As[buf][r * LS + c], 0, 0);
#else
      regA[i] = *(const unsigned long long*)&Au[(size_t)r * K + k0 + c];
#endif
    }
#pragma unroll
    for (int i = 0; i < 8; ++i) {
      int idx = (i * T + tid) * 4;
      int r = idx >> 5, c = idx & 31;
#if USE_ASYNC_LDS
      __builtin_amdgcn_global_load_async_to_lds_b64(
          (AS_GLOBAL v2i_t*)(Bu + (size_t)r * K + k0 + c),
          (AS_LOCAL v2i_t*)&Bs[buf][r * LS + c], 0, 0);
#else
      regB[i] = *(const unsigned long long*)&Bu[(size_t)r * K + k0 + c];
#endif
    }
  };

  // Make buffer `buf` contain the issued tile (wave-local part)
  auto commit = [&](int buf) {
#if USE_ASYNC_LDS
    __builtin_amdgcn_s_wait_asynccnt(0);
    (void)buf;
#else
#pragma unroll
    for (int i = 0; i < NCA; ++i) {
      int idx = (i * T + tid) * 4;
      int r = idx >> 5, c = idx & 31;
      *(unsigned long long*)&As[buf][r * LS + c] = regA[i];
    }
#pragma unroll
    for (int i = 0; i < 8; ++i) {
      int idx = (i * T + tid) * 4;
      int r = idx >> 5, c = idx & 31;
      *(unsigned long long*)&Bs[buf][r * LS + c] = regB[i];
    }
#endif
  };

  // 8 WMMAs on LDS buffer `buf`
  auto compute = [&](int buf) {
    const int ml = lane & 15;
    Frag16 a0, a1;
    {
      // ISA 16-bit A layout: lane L: M=L, K{0..7,16..23}; lane L+16: K{8..15,24..31}
      const int kh = (lane >> 4) * 8;
#pragma unroll
      for (int v = 0; v < 8; ++v) {
        int k = ((v < 4) ? (2 * v) : (2 * v + 8)) + kh;
        a0.u[v] = *(const unsigned*)&As[buf][ml * LS + k];
        a1.u[v] = *(const unsigned*)&As[buf][(ml + 16) * LS + k];
      }
    }
    // ISA 16-bit B layout: lane = one N column; lanes 0-15 K{0..7,16..23}, 16-31 K{8..15,24..31}
    const int kb = (lane >> 4) * 8;
#pragma unroll
    for (int j = 0; j < 4; ++j) {
      Frag16 bf;
      const int nrow = waveN * 64 + j * 16 + ml;
#pragma unroll
      for (int v = 0; v < 8; ++v)
        bf.u[v] = *(const unsigned*)&Bs[buf][nrow * LS + ((v < 4) ? (2 * v) : (2 * v + 8)) + kb];
      acc[0][j] = __builtin_amdgcn_wmma_f32_16x16x32_f16(
          false, a0.v, false, bf.v, (short)0, acc[0][j], false, false);
      acc[1][j] = __builtin_amdgcn_wmma_f32_16x16x32_f16(
          false, a1.v, false, bf.v, (short)0, acc[1][j], false, false);
    }
  };

  const int nTiles = K / BK;
  issue(0, 0);
  commit(0);
  __syncthreads();
  for (int t = 0; t < nTiles; ++t) {
    const int cur = t & 1;
    if (t + 1 < nTiles) issue(t + 1, cur ^ 1);  // overlap with compute below
    compute(cur);
    if (t + 1 < nTiles) {
      commit(cur ^ 1);
      __syncthreads();
    }
  }

  // Epilogue. C/D layout: VGPR v -> M = v + (lane>>4)*8, N = lane&15.
  const int ml = lane & 15;
  const int mb = blockRow + ((lane >> 4) << 3);
#pragma unroll
  for (int j = 0; j < 4; ++j) {
    const int ncol = blockCol + waveN * 64 + j * 16 + ml;
    const float b = colBias ? colBias[ncol] : 0.f;
#pragma unroll
    for (int i = 0; i < 2; ++i)
#pragma unroll
      for (int v = 0; v < 8; ++v)
        C[(size_t)(mb + i * 16 + v) * N + ncol] = acc[i][j][v] + b;
  }
}

// ---------------------------------------------------------------------------
// Per-node attention coefficients: es[n,h] = <feat[n,h,:], a_src[h,:]>, same dst.
// One wave per node, C=128 fixed, float4 per lane + shuffle reduction.
// ---------------------------------------------------------------------------
template <int H>
__global__ __launch_bounds__(256) void score_kernel(
    const float* __restrict__ feat, const float* __restrict__ a_src,
    const float* __restrict__ a_dst, float* __restrict__ es,
    float* __restrict__ ed, int N) {
  int wave = blockIdx.x * (blockDim.x >> 5) + (threadIdx.x >> 5);
  int lane = threadIdx.x & 31;
  if (wave >= N) return;
  const float4* frow = (const float4*)(feat + (size_t)wave * H * 128);
#pragma unroll
  for (int h = 0; h < H; ++h) {
    float4 f  = frow[h * 32 + lane];
    float4 as = ((const float4*)a_src)[h * 32 + lane];
    float4 ad = ((const float4*)a_dst)[h * 32 + lane];
    float ps = f.x * as.x + f.y * as.y + f.z * as.z + f.w * as.w;
    float pd = f.x * ad.x + f.y * ad.y + f.z * ad.z + f.w * ad.w;
#pragma unroll
    for (int off = 16; off > 0; off >>= 1) {
      ps += __shfl_xor(ps, off);
      pd += __shfl_xor(pd, off);
    }
    if (lane == 0) { es[wave * H + h] = ps; ed[wave * H + h] = pd; }
  }
}

// ---------------------------------------------------------------------------
// Edge softmax passes
// ---------------------------------------------------------------------------
template <int H>
__global__ void edge_max_kernel(const int* __restrict__ ei,
                                const float* __restrict__ es,
                                const float* __restrict__ ed,
                                unsigned* __restrict__ m_ord, int E) {
  int t = blockIdx.x * blockDim.x + threadIdx.x;
  if (t >= E * H) return;
  int e = t / H, h = t - e * H;
  int s = ei[e], d = ei[E + e];
  float x = es[s * H + h] + ed[d * H + h];
  x = x > 0.f ? x : 0.2f * x;  // leaky_relu(0.2)
  atomicMax(&m_ord[d * H + h], f2ord(x));
}

template <int H>
__global__ void edge_expsum_kernel(const int* __restrict__ ei,
                                   const float* __restrict__ es,
                                   const float* __restrict__ ed,
                                   const unsigned* __restrict__ m_ord,
                                   float* __restrict__ ssum, int E) {
  int t = blockIdx.x * blockDim.x + threadIdx.x;
  if (t >= E * H) return;
  int e = t / H, h = t - e * H;
  int s = ei[e], d = ei[E + e];
  float x = es[s * H + h] + ed[d * H + h];
  x = x > 0.f ? x : 0.2f * x;
  float m = ord2f(m_ord[d * H + h]);
  atomic_fadd(&ssum[d * H + h], expf(x - m));
}

// One wave per edge; lanes cover H*128 features with float4 gathers from
// feat[src] and hardware fp32 scatter-adds into out[dst] (init'd to bias).
template <int H>
__global__ __launch_bounds__(256) void edge_aggregate_kernel(
    const int* __restrict__ ei, const float* __restrict__ feat,
    const float* __restrict__ es, const float* __restrict__ ed,
    const unsigned* __restrict__ m_ord, const float* __restrict__ ssum,
    float* __restrict__ out, int E) {
  int wave = blockIdx.x * (blockDim.x >> 5) + (threadIdx.x >> 5);
  int lane = threadIdx.x & 31;
  if (wave >= E) return;
  int s = ei[wave], d = ei[E + wave];
  const float4* fsrc = (const float4*)(feat + (size_t)s * H * 128);
  float* odst = out + (size_t)d * H * 128;
#pragma unroll
  for (int h = 0; h < H; ++h) {
    float x = es[s * H + h] + ed[d * H + h];
    x = x > 0.f ? x : 0.2f * x;
    float m = ord2f(m_ord[d * H + h]);
    float alpha = expf(x - m) / (ssum[d * H + h] + 1e-16f);
    float4 f = fsrc[h * 32 + lane];
    int base = h * 128 + lane * 4;
    atomic_fadd(&odst[base + 0], f.x * alpha);
    atomic_fadd(&odst[base + 1], f.y * alpha);
    atomic_fadd(&odst[base + 2], f.z * alpha);
    atomic_fadd(&odst[base + 3], f.w * alpha);
  }
}

// ELU then convert to f16 (input of GEMM2)
__global__ void elu_f16_kernel(const float* __restrict__ in,
                               f16_t* __restrict__ out, size_t n) {
  size_t t = (size_t)blockIdx.x * blockDim.x + threadIdx.x;
  if (t >= n) return;
  float v = in[t];
  v = v > 0.f ? v : (expf(v) - 1.f);
  out[t] = (f16_t)v;
}

// ---------------------------------------------------------------------------
// Host-side orchestration
// ---------------------------------------------------------------------------
extern "C" void kernel_launch(void* const* d_in, const int* in_sizes, int n_in,
                              void* d_out, int out_size, void* d_ws, size_t ws_size,
                              hipStream_t stream) {
  const float* x    = (const float*)d_in[0];
  const int*   ei   = (const int*)d_in[1];
  const float* sent = (const float*)d_in[2];
  const float* W1   = (const float*)d_in[3];
  const float* a1s  = (const float*)d_in[4];
  const float* a1d  = (const float*)d_in[5];
  const float* b1   = (const float*)d_in[6];
  const float* W2   = (const float*)d_in[7];
  const float* a2s  = (const float*)d_in[8];
  const float* a2d  = (const float*)d_in[9];
  const float* b2   = (const float*)d_in[10];
  float* out = (float*)d_out;
  (void)n_in; (void)out_size; (void)ws_size;

  const int SED = 768, HC1 = 512, C = 128;
  const int N = in_sizes[0] / SED;   // 20000
  const int E = in_sizes[1] / 2;     // 320000

  char* ws = (char*)d_ws;
  size_t off = 0;
  auto alloc = [&](size_t bytes) -> char* {
    char* p = ws + off;
    off = (off + bytes + 255) & ~(size_t)255;
    return p;
  };
  f16_t*    xb   = (f16_t*)alloc((size_t)N * SED * 2);
  f16_t*    w1t  = (f16_t*)alloc((size_t)HC1 * SED * 2);  // [512,768]
  f16_t*    w2t  = (f16_t*)alloc((size_t)C * HC1 * 2);    // [128,512]
  float*    sp   = (float*)alloc(HC1 * 4);
  float*    h1   = (float*)alloc((size_t)N * HC1 * 4);
  float*    es1  = (float*)alloc((size_t)N * 4 * 4);
  float*    ed1  = (float*)alloc((size_t)N * 4 * 4);
  unsigned* m1   = (unsigned*)alloc((size_t)N * 4 * 4);
  float*    s1   = (float*)alloc((size_t)N * 4 * 4);
  float*    out1 = (float*)alloc((size_t)N * HC1 * 4);
  f16_t*    h1b  = (f16_t*)alloc((size_t)N * HC1 * 2);
  float*    h2   = (float*)alloc((size_t)N * C * 4);
  float*    es2  = (float*)alloc((size_t)N * 4);
  float*    ed2  = (float*)alloc((size_t)N * 4);
  unsigned* m2   = (unsigned*)alloc((size_t)N * 4);
  float*    s2   = (float*)alloc((size_t)N * 4);

  // ---- precision conversion & weight transposes ----
  {
    size_t n = (size_t)N * SED;
    f32_to_f16_kernel<<<(unsigned)((n + 255) / 256), 256, 0, stream>>>(x, xb, n);
  }
  transpose_f16_kernel<<<(SED * HC1 + 255) / 256, 256, 0, stream>>>(W1, w1t, SED, HC1);
  transpose_f16_kernel<<<(HC1 * C + 255) / 256, 256, 0, stream>>>(W2, w2t, HC1, C);
  sent_proj_kernel<<<2, 256, 0, stream>>>(sent, W1, sp);

  // ---- init accumulators: out buffers start at bias; softmax state zeroed ----
  fill_bias_kernel<<<(N * HC1 + 255) / 256, 256, 0, stream>>>(out1, b1, N, HC1);
  fill_bias_kernel<<<(N * C + 255) / 256, 256, 0, stream>>>(out, b2, N, C);
  fill_ms_kernel<<<(N * 4 + 255) / 256, 256, 0, stream>>>(m1, s1, N * 4);
  fill_ms_kernel<<<(N + 255) / 256, 256, 0, stream>>>(m2, s2, N);

  // ---- layer 1: h1 = x @ W1[:768] + sent@W1[768:]  (WMMA, BN=512) ----
  gemm_f16_wmma_kernel<8><<<dim3(N / 32, HC1 / 512), 256, 0, stream>>>(
      xb, w1t, h1, sp, N, HC1, SED);
  score_kernel<4><<<(N + 7) / 8, 256, 0, stream>>>(h1, a1s, a1d, es1, ed1, N);
  edge_max_kernel<4><<<(E * 4 + 255) / 256, 256, 0, stream>>>(ei, es1, ed1, m1, E);
  edge_expsum_kernel<4><<<(E * 4 + 255) / 256, 256, 0, stream>>>(ei, es1, ed1, m1, s1, E);
  edge_aggregate_kernel<4><<<(E + 7) / 8, 256, 0, stream>>>(ei, h1, es1, ed1, m1, s1, out1, E);
  {
    size_t n = (size_t)N * HC1;
    elu_f16_kernel<<<(unsigned)((n + 255) / 256), 256, 0, stream>>>(out1, h1b, n);
  }

  // ---- layer 2: h2 = elu(out1) @ W2  (WMMA, BN=128), heads=1 ----
  gemm_f16_wmma_kernel<2><<<dim3(N / 32, C / 128), 64, 0, stream>>>(
      h1b, w2t, h2, nullptr, N, C, HC1);
  score_kernel<1><<<(N + 7) / 8, 256, 0, stream>>>(h2, a2s, a2d, es2, ed2, N);
  edge_max_kernel<1><<<(E + 255) / 256, 256, 0, stream>>>(ei, es2, ed2, m2, E);
  edge_expsum_kernel<1><<<(E + 255) / 256, 256, 0, stream>>>(ei, es2, ed2, m2, s2, E);
  edge_aggregate_kernel<1><<<(E + 7) / 8, 256, 0, stream>>>(ei, h2, es2, ed2, m2, s2, out, E);
}
